// AttentiveConvLSTM_25262997635269
// MI455X (gfx1250) — matrix operand order repaired and checked by reading, not compiled, not run on hardware
//
#include <hip/hip_runtime.h>

typedef __attribute__((ext_vector_type(16))) _Float16 v16h;
typedef __attribute__((ext_vector_type(8)))  _Float16 v8h;
typedef __attribute__((ext_vector_type(8)))  float    v8f;

#define Bn   4
#define Tn   4
#define Hh   30
#define Ww   40
#define HW   1200
#define Mtot 4800          // B*H*W
#define C1   256
#define C2   512           // concat channels
#define K9   (9*C2)        // 4608 = GEMM K

__device__ __forceinline__ float sigm(float x) { return 1.f / (1.f + __expf(-x)); }

// Build a 16x32 f16 A-fragment from two 16B chunks (contiguous layout:
// lane l row m=l&15; K=e+16*(l>>4)).
__device__ __forceinline__ v16h make_afrag(const _Float16* __restrict__ p) {
  v8h lo = *(const v8h*)(p);
  v8h hi = *(const v8h*)(p + 8);
  v16h a;
#pragma unroll
  for (int e = 0; e < 8; ++e) { a[e] = lo[e]; a[e + 8] = hi[e]; }
  return a;
}

// ---------------------------------------------------------------- GEMM: attention pre-act
// z[b][hw][n] = tanh( conv([h;xt]) + bias ), f16 NHWC out. 2 m-tiles per wave.
__global__ __launch_bounds__(256) void gemm_att_kernel(
    const _Float16* __restrict__ act, const _Float16* __restrict__ wp,
    const float* __restrict__ bias, const _Float16* __restrict__ zpad,
    _Float16* __restrict__ z) {
  int wave  = (int)((blockIdx.x * blockDim.x + threadIdx.x) >> 5);
  int lane  = threadIdx.x & 31;
  int mPair = wave >> 4;            // 150 pairs of m-tiles
  int nTile = wave & 15;            // N = 256 -> 16 n-tiles
  int lm = lane & 15, khalf = lane >> 4;
  int n  = nTile * 16 + lm;

  int m0 = mPair * 32 + lm, m1 = m0 + 16;
  int b0 = m0 / HW, hw0 = m0 % HW, hh0 = hw0 / Ww, ww0 = hw0 % Ww;
  int b1 = m1 / HW, hw1 = m1 % HW, hh1 = hw1 / Ww, ww1 = hw1 % Ww;

  const _Float16* pBn = wp + (size_t)n * K9 + 16 * khalf;   // B: lane K-column
  const _Float16* zp  = zpad + 16 * khalf;

  v8f acc0 = {}, acc1 = {};
  for (int tap = 0; tap < 9; ++tap) {
    int dh = tap / 3 - 1, dw = tap % 3 - 1;
    int hs0 = hh0 + dh, ws0 = ww0 + dw;
    int hs1 = hh1 + dh, ws1 = ww1 + dw;
    bool in0 = ((unsigned)hs0 < Hh) & ((unsigned)ws0 < Ww);
    bool in1 = ((unsigned)hs1 < Hh) & ((unsigned)ws1 < Ww);
    const _Float16* pa0 =
        in0 ? act + (((size_t)b0 * Hh + hs0) * Ww + ws0) * C2 + 16 * khalf : zp;
    const _Float16* pa1 =
        in1 ? act + (((size_t)b1 * Hh + hs1) * Ww + ws1) * C2 + 16 * khalf : zp;
    const _Float16* pb = pBn + (size_t)tap * C2;
#pragma unroll
    for (int ct = 0; ct < 16; ++ct) {                 // 16 k-tiles of 32 per tap
      v16h a0 = make_afrag(pa0 + ct * 32);
      v16h a1 = make_afrag(pa1 + ct * 32);
      v16h bf = *(const v16h*)(pb + ct * 32);
      acc0 = __builtin_amdgcn_wmma_f32_16x16x32_f16(false, a0, false, bf, (short)0, acc0, false, false);
      acc1 = __builtin_amdgcn_wmma_f32_16x16x32_f16(false, a1, false, bf, (short)0, acc1, false, false);
    }
  }
  float bnv = bias[n];
#pragma unroll
  for (int r = 0; r < 8; ++r) {
    int m  = mPair * 32 + r + 8 * khalf;
    int b_ = m / HW, hw = m % HW;
    z[((size_t)b_ * HW + hw) * C1 + n] = (_Float16)tanhf(acc0[r] + bnv);
    m  = m + 16;
    b_ = m / HW; hw = m % HW;
    z[((size_t)b_ * HW + hw) * C1 + n] = (_Float16)tanhf(acc1[r] + bnv);
  }
}

// ---------------------------------------------------------------- GEMM: gates + LSTM update
__device__ __forceinline__ void lstm_store(v8f ai, v8f af, v8f ac, v8f ao,
                                           int mBase, int khalf, int n,
                                           float bi, float bf, float bc, float bo,
                                           float* __restrict__ cbuf,
                                           float* __restrict__ hbuf) {
#pragma unroll
  for (int r = 0; r < 8; ++r) {
    int m  = mBase + r + 8 * khalf;
    int b_ = m / HW, hw = m % HW;
    size_t idx = ((size_t)b_ * C1 + n) * HW + hw;   // NCHW f32
    float ig = sigm(ai[r] + bi);
    float fg = sigm(af[r] + bf);
    float gg = tanhf(ac[r] + bc);
    float og = sigm(ao[r] + bo);
    float cn = fg * cbuf[idx] + ig * gg;
    cbuf[idx] = cn;
    hbuf[idx] = og * tanhf(cn);
  }
}

// 2 m-tiles x 4 gates = 8 accumulators per wave; B-fragments reused by both
// A-fragments; full LSTM pointwise update fused in the epilogue.
__global__ __launch_bounds__(256) void gemm_gates_kernel(
    const _Float16* __restrict__ act, const _Float16* __restrict__ wp,
    const float* __restrict__ bias, const _Float16* __restrict__ zpad,
    float* __restrict__ cbuf, float* __restrict__ hbuf) {
  int wave  = (int)((blockIdx.x * blockDim.x + threadIdx.x) >> 5);
  int lane  = threadIdx.x & 31;
  int mPair = wave >> 4;            // 150
  int nTile = wave & 15;            // 16 n-tiles per gate (gate N = 256)
  int lm = lane & 15, khalf = lane >> 4;
  int n  = nTile * 16 + lm;

  int m0 = mPair * 32 + lm, m1 = m0 + 16;
  int b0 = m0 / HW, hw0 = m0 % HW, hh0 = hw0 / Ww, ww0 = hw0 % Ww;
  int b1 = m1 / HW, hw1 = m1 % HW, hh1 = hw1 / Ww, ww1 = hw1 % Ww;

  const _Float16* pBn = wp + (size_t)n * K9 + 16 * khalf;
  const _Float16* zp  = zpad + 16 * khalf;

  v8f ai0 = {}, af0 = {}, ac0 = {}, ao0 = {};
  v8f ai1 = {}, af1 = {}, ac1 = {}, ao1 = {};

  for (int tap = 0; tap < 9; ++tap) {
    int dh = tap / 3 - 1, dw = tap % 3 - 1;
    int hs0 = hh0 + dh, ws0 = ww0 + dw;
    int hs1 = hh1 + dh, ws1 = ww1 + dw;
    bool in0 = ((unsigned)hs0 < Hh) & ((unsigned)ws0 < Ww);
    bool in1 = ((unsigned)hs1 < Hh) & ((unsigned)ws1 < Ww);
    const _Float16* pa0 =
        in0 ? act + (((size_t)b0 * Hh + hs0) * Ww + ws0) * C2 + 16 * khalf : zp;
    const _Float16* pa1 =
        in1 ? act + (((size_t)b1 * Hh + hs1) * Ww + ws1) * C2 + 16 * khalf : zp;
    const _Float16* pb = pBn + (size_t)tap * C2;
#pragma unroll
    for (int ct = 0; ct < 16; ++ct) {
      v16h a0 = make_afrag(pa0 + ct * 32);
      v16h a1 = make_afrag(pa1 + ct * 32);
      // gate g weights live at n + 256*g -> constant offset g*C1*K9 elements
      v16h bi_ = *(const v16h*)(pb + ct * 32);
      v16h bf_ = *(const v16h*)(pb + (size_t)1 * C1 * K9 + ct * 32);
      v16h bc_ = *(const v16h*)(pb + (size_t)2 * C1 * K9 + ct * 32);
      v16h bo_ = *(const v16h*)(pb + (size_t)3 * C1 * K9 + ct * 32);
      ai0 = __builtin_amdgcn_wmma_f32_16x16x32_f16(false, a0, false, bi_, (short)0, ai0, false, false);
      ai1 = __builtin_amdgcn_wmma_f32_16x16x32_f16(false, a1, false, bi_, (short)0, ai1, false, false);
      af0 = __builtin_amdgcn_wmma_f32_16x16x32_f16(false, a0, false, bf_, (short)0, af0, false, false);
      af1 = __builtin_amdgcn_wmma_f32_16x16x32_f16(false, a1, false, bf_, (short)0, af1, false, false);
      ac0 = __builtin_amdgcn_wmma_f32_16x16x32_f16(false, a0, false, bc_, (short)0, ac0, false, false);
      ac1 = __builtin_amdgcn_wmma_f32_16x16x32_f16(false, a1, false, bc_, (short)0, ac1, false, false);
      ao0 = __builtin_amdgcn_wmma_f32_16x16x32_f16(false, a0, false, bo_, (short)0, ao0, false, false);
      ao1 = __builtin_amdgcn_wmma_f32_16x16x32_f16(false, a1, false, bo_, (short)0, ao1, false, false);
    }
  }
  float bi = bias[n], bf = bias[n + 256], bc = bias[n + 512], bo = bias[n + 768];
  lstm_store(ai0, af0, ac0, ao0, mPair * 32,      khalf, n, bi, bf, bc, bo, cbuf, hbuf);
  lstm_store(ai1, af1, ac1, ao1, mPair * 32 + 16, khalf, n, bi, bf, bc, bo, cbuf, hbuf);
}

// ---------------------------------------------------------------- small ops

__global__ __launch_bounds__(256) void econv_kernel(const _Float16* __restrict__ z,
                                                    const float* __restrict__ Va,
                                                    float* __restrict__ e) {
  int pos = blockIdx.x;            // 4800 positions
  int b = pos / HW, hw = pos % HW;
  int hh = hw / Ww, ww = hw % Ww;
  int ci = threadIdx.x;            // 256 channels
  float s = 0.f;
#pragma unroll
  for (int tap = 0; tap < 9; ++tap) {
    int hs = hh + tap / 3 - 1, wsd = ww + tap % 3 - 1;
    if (hs < 0 || hs >= Hh || wsd < 0 || wsd >= Ww) continue;
    s += (float)z[(((size_t)b * Hh + hs) * Ww + wsd) * C1 + ci] * Va[tap * C1 + ci];
  }
  __shared__ float red[256];
  red[ci] = s; __syncthreads();
  for (int off = 128; off > 0; off >>= 1) {
    if (ci < off) red[ci] += red[ci + off];
    __syncthreads();
  }
  if (ci == 0) e[pos] = red[0];
}

__global__ __launch_bounds__(256) void softmax_kernel(float* __restrict__ ea) {
  float* p = ea + (size_t)blockIdx.x * HW;
  __shared__ float red[256];
  __shared__ float buf[HW];
  int tid = threadIdx.x;
  float mx = -1e30f;
  for (int i = tid; i < HW; i += 256) mx = fmaxf(mx, p[i]);
  red[tid] = mx; __syncthreads();
  for (int off = 128; off > 0; off >>= 1) {
    if (tid < off) red[tid] = fmaxf(red[tid], red[tid + off]);
    __syncthreads();
  }
  mx = red[0]; __syncthreads();
  float s = 0.f;
  for (int i = tid; i < HW; i += 256) { float v = __expf(p[i] - mx); buf[i] = v; s += v; }
  red[tid] = s; __syncthreads();
  for (int off = 128; off > 0; off >>= 1) {
    if (tid < off) red[tid] += red[tid + off];
    __syncthreads();
  }
  float inv = 1.f / red[0];
  for (int i = tid; i < HW; i += 256) p[i] = buf[i] * inv;
}

// stage NHWC f16 concat activations.  mode 0: [h ; xt]   mode 1: [xt*a ; h]
__global__ __launch_bounds__(256) void stage_kernel(const float* __restrict__ x, int t,
                                                    const float* __restrict__ h,
                                                    const float* __restrict__ a,
                                                    _Float16* __restrict__ act, int mode) {
  size_t idx = (size_t)blockIdx.x * blockDim.x + threadIdx.x;
  if (idx >= (size_t)Mtot * C2) return;
  int ci = (int)(idx & 511);
  int m  = (int)(idx >> 9);
  int b = m / HW, hw = m % HW;
  const float* xt = x + ((size_t)b * Tn + t) * (size_t)C1 * HW;
  float v;
  if (mode == 0)
    v = (ci < C1) ? h[((size_t)b * C1 + ci) * HW + hw]
                  : xt[(size_t)(ci - C1) * HW + hw];
  else
    v = (ci < C1) ? xt[(size_t)ci * HW + hw] * a[(size_t)b * HW + hw]
                  : h[((size_t)b * C1 + (ci - C1)) * HW + hw];
  act[idx] = (_Float16)v;
}

__global__ __launch_bounds__(256) void init_kernel(const float* __restrict__ x,
                                                   float* __restrict__ h,
                                                   float* __restrict__ c) {
  size_t idx = (size_t)blockIdx.x * blockDim.x + threadIdx.x;
  if (idx >= (size_t)Bn * C1 * HW) return;
  size_t b = idx / ((size_t)C1 * HW);
  size_t r = idx % ((size_t)C1 * HW);
  float s = 0.f;
#pragma unroll
  for (int t = 0; t < Tn; ++t) s += x[(b * Tn + t) * (size_t)C1 * HW + r];
  h[idx] = s; c[idx] = s;
}

__global__ __launch_bounds__(256) void pack_att_kernel(const float* __restrict__ Wa,
                                                       const float* __restrict__ Ua,
                                                       _Float16* __restrict__ wp) {
  size_t idx = (size_t)blockIdx.x * blockDim.x + threadIdx.x;
  if (idx >= (size_t)C1 * K9) return;
  int k = (int)(idx % K9);
  int n = (int)(idx / K9);
  int tap = k / C2, ci = k % C2;
  float v = (ci < C1) ? Wa[((size_t)tap * C1 + ci) * C1 + n]
                      : Ua[((size_t)tap * C1 + (ci - C1)) * C1 + n];
  wp[idx] = (_Float16)v;
}

__global__ __launch_bounds__(256) void pack_g_kernel(const float* __restrict__ Wx,
                                                     const float* __restrict__ Uh,
                                                     _Float16* __restrict__ wp) {
  size_t idx = (size_t)blockIdx.x * blockDim.x + threadIdx.x;
  if (idx >= (size_t)1024 * K9) return;
  int k = (int)(idx % K9);
  int n = (int)(idx / K9);
  int tap = k / C2, ci = k % C2;
  float v = (ci < C1) ? Wx[((size_t)tap * C1 + ci) * 1024 + n]
                      : Uh[((size_t)tap * C1 + (ci - C1)) * 1024 + n];
  wp[idx] = (_Float16)v;
}

__global__ __launch_bounds__(256) void pack_bias_kernel(const float* ba, const float* bua,
                                                        const float* bx, const float* bh,
                                                        float* batt, float* bg,
                                                        _Float16* zpad) {
  int i = blockIdx.x * blockDim.x + threadIdx.x;
  if (i < C1)   batt[i] = ba[i] + bua[i];
  if (i < 1024) { bg[i] = bx[i] + bh[i]; zpad[i] = (_Float16)0.f; }
}

// ---------------------------------------------------------------- host

extern "C" void kernel_launch(void* const* d_in, const int* in_sizes, int n_in,
                              void* d_out, int out_size, void* d_ws, size_t ws_size,
                              hipStream_t stream) {
  const float* x   = (const float*)d_in[0];
  const float* Wa  = (const float*)d_in[1];
  const float* ba  = (const float*)d_in[2];
  const float* Ua  = (const float*)d_in[3];
  const float* bua = (const float*)d_in[4];
  const float* Va  = (const float*)d_in[5];
  const float* Wx  = (const float*)d_in[6];
  const float* bx  = (const float*)d_in[7];
  const float* Uh  = (const float*)d_in[8];
  const float* bh  = (const float*)d_in[9];

  char* ws = (char*)d_ws;
  size_t off = 0;
  auto alloc = [&](size_t bytes) -> void* {
    void* p = ws + off;
    off = (off + bytes + 255) & ~(size_t)255;
    return p;
  };
  _Float16* wp_att = (_Float16*)alloc((size_t)C1 * K9 * 2);        // 2.36 MB
  _Float16* wp_g   = (_Float16*)alloc((size_t)1024 * K9 * 2);      // 9.44 MB
  float*    batt   = (float*)alloc(C1 * 4);
  float*    bg     = (float*)alloc(1024 * 4);
  _Float16* zpad   = (_Float16*)alloc(1024 * 2);                   // zero page for OOB taps
  _Float16* act    = (_Float16*)alloc((size_t)Mtot * C2 * 2);      // 4.92 MB
  _Float16* zbuf   = (_Float16*)alloc((size_t)Mtot * C1 * 2);      // 2.46 MB
  float*    ea     = (float*)alloc((size_t)Mtot * 4);
  float*    hbuf   = (float*)alloc((size_t)Bn * C1 * HW * 4);      // NCHW
  float*    cbuf   = (float*)alloc((size_t)Bn * C1 * HW * 4);      // NCHW

  pack_att_kernel<<<((size_t)C1 * K9 + 255) / 256, 256, 0, stream>>>(Wa, Ua, wp_att);
  pack_g_kernel<<<((size_t)1024 * K9 + 255) / 256, 256, 0, stream>>>(Wx, Uh, wp_g);
  pack_bias_kernel<<<4, 256, 0, stream>>>(ba, bua, bx, bh, batt, bg, zpad);
  init_kernel<<<((size_t)Bn * C1 * HW + 255) / 256, 256, 0, stream>>>(x, hbuf, cbuf);

  const int gemmBlocks = (Mtot / 32) * 16 / 8;   // 150 m-pairs * 16 n-tiles / 8 waves
  for (int t = 0; t < Tn; ++t) {
    stage_kernel<<<((size_t)Mtot * C2 + 255) / 256, 256, 0, stream>>>(x, t, hbuf, ea, act, 0);
    gemm_att_kernel<<<gemmBlocks, 256, 0, stream>>>(act, wp_att, batt, zpad, zbuf);
    econv_kernel<<<Mtot, 256, 0, stream>>>(zbuf, Va, ea);
    softmax_kernel<<<Bn, 256, 0, stream>>>(ea);
    stage_kernel<<<((size_t)Mtot * C2 + 255) / 256, 256, 0, stream>>>(x, t, hbuf, ea, act, 1);
    gemm_gates_kernel<<<gemmBlocks, 256, 0, stream>>>(act, wp_g, bg, zpad, cbuf, hbuf);
  }
  hipMemcpyAsync(d_out, hbuf, (size_t)Bn * C1 * HW * 4, hipMemcpyDeviceToDevice, stream);
}
